// SO3FrameDenoisingLayer_20117626814828
// MI455X (gfx1250) — matrix operand order repaired and checked by reading, not compiled, not run on hardware
//
#include <hip/hip_runtime.h>
#include <hip/hip_bf16.h>
#include <math.h>

#define NND 20000
#define NE  600000
#define NSE 40000

typedef __attribute__((ext_vector_type(16))) __bf16 v16bf;
typedef __attribute__((ext_vector_type(8)))  float  v8f;

__device__ __forceinline__ float siluf(float x){ return x / (1.f + __expf(-x)); }
__device__ __forceinline__ float sigf (float x){ return 1.f / (1.f + __expf(-x)); }

// pack two f32 into one u32 of bf16 pairs (lo = first)
__device__ __forceinline__ unsigned int pk2(float a, float b){
  union { __bf16 h[2]; unsigned int u; } r;
  r.h[0] = (__bf16)a; r.h[1] = (__bf16)b;
  return r.u;
}
__device__ __forceinline__ float bflo(unsigned int u){
  union { unsigned short s; __bf16 h; } r; r.s=(unsigned short)(u&0xFFFFu); return (float)r.h;
}
__device__ __forceinline__ float bfhi(unsigned int u){
  union { unsigned short s; __bf16 h; } r; r.s=(unsigned short)(u>>16); return (float)r.h;
}

// ---- WMMA operand loaders ---------------------------------------------------
// A slab 16x32: lane m=L&15, half=L>>4; VGPR q holds K=(q>>2)*16+half*8+(q&3)*2 (+0,+1)
__device__ __forceinline__ v16bf ldA(const float* __restrict__ A, int lda, int kbase){
  int lane = threadIdx.x & 31;
  const float* row = A + (lane & 15)*lda + kbase + (lane >> 4)*8;
  v16bf a;
#pragma unroll
  for (int q = 0; q < 8; ++q){
    int k = (q>>2)*16 + (q&3)*2;
    a[2*q]   = (__bf16)row[k];
    a[2*q+1] = (__bf16)row[k+1];
  }
  return a;
}
// A from bf16-pair-packed LDS (lda in elements, even): 2x ds_load_b128 per slab
__device__ __forceinline__ v16bf ldAh(const unsigned int* __restrict__ A2, int lda, int kbase){
  int lane = threadIdx.x & 31;
  const unsigned int* row = A2 + (lane & 15)*(lda>>1) + ((kbase + (lane>>4)*8)>>1);
  union { v16bf b; unsigned int u[8]; } r;
#pragma unroll
  for (int q = 0; q < 8; ++q)
    r.u[q] = row[(q>>2)*8 + (q&3)];
  return r.b;
}
// B slab 32x16 from swizzled weights [ntile][kslab][lane][q]: lane's 8 words are
// contiguous (32B) -> 2x global_load_b128 per slab.
__device__ __forceinline__ v16bf ldBh(const unsigned int* __restrict__ Wp, int K, int kbase, int ncol){
  int lane = threadIdx.x & 31;
  const unsigned int* base = Wp + ((((size_t)(ncol>>4)*(K>>5)) + (kbase>>5))*32 + lane)*8;
  union { v16bf b; unsigned int u[8]; } r;
#pragma unroll
  for (int q = 0; q < 8; ++q) r.u[q] = base[q];
  return r.b;
}

// ---- GEMM tile drivers ------------------------------------------------------
__device__ __forceinline__ v8f gemmH(const unsigned int* A2, int lda,
                                     const unsigned int* Wp, int K, int ncol){
  v8f acc = {0.f,0.f,0.f,0.f,0.f,0.f,0.f,0.f};
  for (int kb = 0; kb < K; kb += 32){
    v16bf a = ldAh(A2, lda, kb);
    v16bf b = ldBh(Wp, K, kb, ncol);
    acc = __builtin_amdgcn_wmma_f32_16x16x32_bf16(false, a, false, b, (short)0, acc, false, false);
  }
  return acc;
}
__device__ __forceinline__ void gemmH2(const unsigned int* A2, int lda,
                                       const unsigned int* Wp, int K,
                                       int n0, int n1, v8f& acc0, v8f& acc1){
  for (int kb = 0; kb < K; kb += 32){
    v16bf a  = ldAh(A2, lda, kb);
    v16bf b0 = ldBh(Wp, K, kb, n0);
    v16bf b1 = ldBh(Wp, K, kb, n1);
    acc0 = __builtin_amdgcn_wmma_f32_16x16x32_bf16(false, a, false, b0, (short)0, acc0, false, false);
    acc1 = __builtin_amdgcn_wmma_f32_16x16x32_bf16(false, a, false, b1, (short)0, acc1, false, false);
  }
}
__device__ __forceinline__ v8f gemmF(const float* A, int lda,
                                     const unsigned int* Wp, int K, int ncol){
  v8f acc = {0.f,0.f,0.f,0.f,0.f,0.f,0.f,0.f};
  for (int kb = 0; kb < K; kb += 32){
    v16bf a = ldA(A, lda, kb);
    v16bf b = ldBh(Wp, K, kb, ncol);
    acc = __builtin_amdgcn_wmma_f32_16x16x32_bf16(false, a, false, b, (short)0, acc, false, false);
  }
  return acc;
}
__device__ __forceinline__ void gemmF2(const float* A, int lda,
                                       const unsigned int* Wp, int K,
                                       int n0, int n1, v8f& acc0, v8f& acc1){
  for (int kb = 0; kb < K; kb += 32){
    v16bf a  = ldA(A, lda, kb);
    v16bf b0 = ldBh(Wp, K, kb, n0);
    v16bf b1 = ldBh(Wp, K, kb, n1);
    acc0 = __builtin_amdgcn_wmma_f32_16x16x32_bf16(false, a, false, b0, (short)0, acc0, false, false);
    acc1 = __builtin_amdgcn_wmma_f32_16x16x32_bf16(false, a, false, b1, (short)0, acc1, false, false);
  }
}
#define VZERO {0.f,0.f,0.f,0.f,0.f,0.f,0.f,0.f}

// D element (vgpr r, lane L) = D[(L>>4)*8 + r][L&15]
template<int ACT, bool ADD>
__device__ __forceinline__ void epi(float* O, int ldo, int ncol, v8f acc, const float* bias){
  int lane = threadIdx.x & 31;
  int n = lane & 15, mo = (lane>>4)*8;
  float bv = bias ? bias[ncol+n] : 0.f;
#pragma unroll
  for (int r=0;r<8;++r){
    float x = acc[r] + bv;
    if (ACT==1) x = siluf(x);
    if (ACT==2) x = fmaxf(x, 0.f);
    if (ACT==3) x = sigf(x);
    float* p = O + (mo+r)*ldo + ncol + n;
    if (ADD) *p += x; else *p = x;
  }
}
template<int ACT>
__device__ __forceinline__ void epiH(__bf16* O, int ldo, int ncol, v8f acc, const float* bias){
  int lane = threadIdx.x & 31;
  int n = lane & 15, mo = (lane>>4)*8;
  float bv = bias ? bias[ncol+n] : 0.f;
#pragma unroll
  for (int r=0;r<8;++r){
    float x = acc[r] + bv;
    if (ACT==1) x = siluf(x);
    if (ACT==2) x = fmaxf(x, 0.f);
    O[(mo+r)*ldo + ncol + n] = (__bf16)x;
  }
}
__device__ __forceinline__ void rowstats(const float* row, int n, float& mu, float& rs){
  float s=0.f, ss=0.f;
  for (int i=0;i<n;++i){ float x=row[i]; s+=x; ss+=x*x; }
  mu = s/(float)n;
  rs = rsqrtf(fmaxf(ss/(float)n - mu*mu, 0.f) + 1e-5f);
}

// ---- weight pre-pack: f32 row-major [K][N] -> swizzled bf16 pairs -----------
// word index i = ((ntile*(K/32) + kslab)*32 + lane)*8 + q
// holds pair (k = kslab*32 + (q>>2)*16 + (lane>>4)*8 + (q&3)*2, n = ntile*16 + (lane&15))
__global__ void k_packw(const float* __restrict__ W, unsigned int* __restrict__ out,
                        int total, int K, int N){
  int i=blockIdx.x*blockDim.x+threadIdx.x;
  if (i>=total) return;
  int q = i & 7;
  int L = (i >> 3) & 31;
  int rest = i >> 8;
  int nslab = K >> 5;
  int t = rest / nslab, s2 = rest - t*nslab;
  int k = s2*32 + (q>>2)*16 + (L>>4)*8 + (q&3)*2;
  int n = t*16 + (L&15);
  out[i] = pk2(W[(size_t)k*N+n], W[(size_t)(k+1)*N+n]);
}

__global__ void k_fill(float* __restrict__ p, int n){
  int i=blockIdx.x*blockDim.x+threadIdx.x;
  if (i<n) p[i]=0.f;
}

// ---- init: s = cat(so3[:,0,:], nf) @ cW + cb ;  v = so3[:,1:4,:] ------------
__global__ __launch_bounds__(128) void k_init(
    const float* __restrict__ so3, const float* __restrict__ nodef,
    const unsigned int* __restrict__ cWp, const float* __restrict__ cb,
    float* __restrict__ s, float* __restrict__ v)
{
  __shared__ unsigned int cat2[16*160];
  const int tid=threadIdx.x, w=tid>>5;
  const int n0=blockIdx.x*16;
  for (int i=tid;i<16*32;i+=128){
    int r=i>>5,c2=i&31;
    const float* p=so3+(size_t)(n0+r)*256+2*c2;
    cat2[r*160+c2]=pk2(p[0],p[1]);
  }
  for (int i=tid;i<16*128;i+=128){
    int r=i>>7,c2=i&127;
    const float* p=nodef+(size_t)(n0+r)*256+2*c2;
    cat2[r*160+32+c2]=pk2(p[0],p[1]);
  }
  for (int i=tid;i<16*192;i+=128){ int r=i/192,q=i%192; v[(size_t)(n0+r)*192+q]=so3[(size_t)(n0+r)*256+64+q]; }
  __syncthreads();
  v8f acc=gemmH(cat2,320,cWp,320,w*16);
  int lane=tid&31, cn=lane&15, mo=(lane>>4)*8, col=w*16+cn;
#pragma unroll
  for (int r=0;r<8;++r) s[(size_t)(n0+mo+r)*64+col]=acc[r]+cb[col];
}

// ---- pass A: edge MLPs -> h (bf16 cache), logits, segment max ---------------
__global__ __launch_bounds__(128) void k_edgeA(
    const float* __restrict__ ef, const int* __restrict__ src, const int* __restrict__ dst,
    const float* __restrict__ s,
    const unsigned int* __restrict__ eW1p, const float* __restrict__ eb1,
    const unsigned int* __restrict__ eW2p, const float* __restrict__ eb2,
    const unsigned int* __restrict__ mWp,  const float* __restrict__ mb,
    const float* __restrict__ aW,  const float* __restrict__ ab,
    unsigned int* __restrict__ hbufu, float* __restrict__ logits,
    unsigned int* __restrict__ nmax)
{
  __shared__ unsigned int bufE2[16*64];   // efeat bf16 pairs, later h
  __shared__ unsigned int bufH2[16*64];   // e1 bf16 pairs
  __shared__ unsigned int catp [16*128];  // [s_src|s_dst|e2] bf16 pairs (256 cols)
  __shared__ int sdst[16];
  const int tid=threadIdx.x, w=tid>>5;
  const int e0=blockIdx.x*16;
  if (blockIdx.x+1 < (int)gridDim.x)
    __builtin_prefetch(ef + (size_t)(e0+16)*128 + tid*16, 0, 1);
  for (int i=tid;i<16*64;i+=128){
    int r=i>>6,c2=i&63;
    const float* p=ef+(size_t)(e0+r)*128+2*c2;
    bufE2[i]=pk2(p[0],p[1]);
  }
  for (int i=tid;i<16*32;i+=128){
    int r=i>>5,c2=i&31;
    const float* ps=s+(size_t)src[e0+r]*64+2*c2;
    const float* pd=s+(size_t)dst[e0+r]*64+2*c2;
    catp[r*128+c2]    = pk2(ps[0],ps[1]);
    catp[r*128+32+c2] = pk2(pd[0],pd[1]);
  }
  if (tid<16) sdst[tid]=dst[e0+tid];
  __syncthreads();
  { v8f a0=VZERO, a1=VZERO;
    gemmH2(bufE2,128,eW1p,128,w*16,(w+4)*16,a0,a1);
    epiH<1>((__bf16*)bufH2,128,w*16,a0,eb1);
    epiH<1>((__bf16*)bufH2,128,(w+4)*16,a1,eb1);
  }
  __syncthreads();
  { v8f a0=VZERO, a1=VZERO;
    gemmH2(bufH2,128,eW2p,128,w*16,(w+4)*16,a0,a1);
    epiH<1>(((__bf16*)catp)+128,256,w*16,a0,eb2);
    epiH<1>(((__bf16*)catp)+128,256,(w+4)*16,a1,eb2);
  }
  __syncthreads();
  { v8f a0=VZERO, a1=VZERO;
    gemmH2(catp,256,mWp,256,w*16,(w+4)*16,a0,a1);
    epiH<1>((__bf16*)bufE2,128,w*16,a0,mb);
    epiH<1>((__bf16*)bufE2,128,(w+4)*16,a1,mb);
  }
  __syncthreads();
  for (int i=tid;i<16*64;i+=128){
    int r=i>>6,c2=i&63;
    hbufu[(size_t)(e0+r)*64+c2] = bufE2[i];   // raw bf16-pair copy
  }
  { // logits: 16 edges x 8 heads == 128 threads
    int r=tid>>3, hh=tid&7;
    float lg=ab[hh];
    for (int k2=0;k2<64;++k2){
      unsigned u=bufE2[r*64+k2];
      lg += bflo(u)*aW[(2*k2)*8+hh] + bfhi(u)*aW[(2*k2+1)*8+hh];
    }
    logits[(size_t)(e0+r)*8+hh]=lg;
    unsigned bits=__float_as_uint(lg);
    unsigned enc=(bits&0x80000000u)?~bits:(bits|0x80000000u); // order-preserving
    atomicMax(&nmax[sdst[r]*8+hh], enc);
  }
}

// ---- pass B: exp(logit - max), segment sum ----------------------------------
__global__ void k_edgeB(const float* __restrict__ logits, const int* __restrict__ dst,
                        const unsigned int* __restrict__ nmax,
                        float* __restrict__ ssum, float* __restrict__ expw, int ne)
{
  int i = blockIdx.x*blockDim.x + threadIdx.x;
  if (i >= ne*8) return;
  int e=i>>3, hh=i&7;
  int d=dst[e];
  unsigned u=nmax[d*8+hh];
  unsigned bits=(u&0x80000000u)?(u^0x80000000u):~u;
  float m=__uint_as_float(bits);
  float ex=__expf(logits[i]-m);
  expw[i]=ex;
  atomicAdd(&ssum[d*8+hh], ex);
}

// ---- pass C: vals/gates GEMMs, rotation, weighted scatter -------------------
__global__ __launch_bounds__(128) void k_edgeC(
    const unsigned int* __restrict__ hbufu,
    const float* __restrict__ expw, const float* __restrict__ ssum,
    const int* __restrict__ src, const int* __restrict__ dst,
    const float* __restrict__ X, const float* __restrict__ v,
    const unsigned int* __restrict__ vWp, const float* __restrict__ vb,
    const unsigned int* __restrict__ gWp, const float* __restrict__ gb,
    float* __restrict__ osacc, float* __restrict__ ovacc)
{
  __shared__ unsigned int bufH2[16*64];  // h bf16 pairs
  __shared__ float bufV[16*128];         // vals, then gates (f32)
  __shared__ float vev[16*192];
  __shared__ float attn[16*8];
  __shared__ float awv[16];
  __shared__ float Rm[16*9];
  __shared__ int ssrc[16], sdst[16];
  const int tid=threadIdx.x, w=tid>>5;
  const int e0=blockIdx.x*16;
  for (int i=tid;i<16*64;i+=128){
    int r=i>>6,c2=i&63;
    bufH2[i]=hbufu[(size_t)(e0+r)*64+c2];
  }
  if (tid<16){ ssrc[tid]=src[e0+tid]; sdst[tid]=dst[e0+tid]; }
  __syncthreads();
  { int r=tid>>3, hh=tid&7;
    attn[tid] = expw[(size_t)(e0+r)*8+hh] / (ssum[sdst[r]*8+hh] + 1e-9f);
  }
  if (tid<16){ // edge rotation matrix, rows [x; n; z]
    int r=tid; int a=ssrc[r], b=sdst[r];
    float dx=X[b*3+0]-X[a*3+0], dy=X[b*3+1]-X[a*3+1], dz=X[b*3+2]-X[a*3+2];
    float il=1.f/(sqrtf(dx*dx+dy*dy+dz*dz)+1e-8f);
    float nx=dx*il, ny=dy*il, nz=dz*il;
    float rx,ry;
    if (fabsf(ny)>0.99f){ rx=1.f; ry=0.f; } else { rx=0.f; ry=1.f; }
    float xx=ry*nz, xy=-rx*nz, xz=rx*ny-ry*nx;            // cross(ref,n), ref.z=0
    float xl=1.f/(sqrtf(xx*xx+xy*xy+xz*xz)+1e-8f);
    xx*=xl; xy*=xl; xz*=xl;
    float zx=ny*xz-nz*xy, zy=nz*xx-nx*xz, zz=nx*xy-ny*xx; // cross(n,x)
    Rm[r*9+0]=xx; Rm[r*9+1]=xy; Rm[r*9+2]=xz;
    Rm[r*9+3]=nx; Rm[r*9+4]=ny; Rm[r*9+5]=nz;
    Rm[r*9+6]=zx; Rm[r*9+7]=zy; Rm[r*9+8]=zz;
  }
  __syncthreads();
  { v8f a0=VZERO, a1=VZERO;
    gemmH2(bufH2,128,vWp,128,w*16,(w+4)*16,a0,a1);
    epi<0,false>(bufV,128,w*16,a0,vb);
    epi<0,false>(bufV,128,(w+4)*16,a1,vb);
  }
  __syncthreads();
  { int c=tid; // scalar message scatter
    for (int r=0;r<16;++r)
      atomicAdd(&osacc[(size_t)sdst[r]*128 + c], attn[r*8 + (c>>4)] * bufV[r*128 + c]);
  }
  if (tid<16){ float s8=0.f; for(int h2=0;h2<8;++h2) s8+=attn[tid*8+h2]; awv[tid]=s8*0.125f; }
  __syncthreads();
  { v8f a0=VZERO, a1=VZERO;
    gemmH2(bufH2,128,gWp,128,w*16,(w+4)*16,a0,a1);
    epi<3,false>(bufV,128,w*16,a0,gb);
    epi<3,false>(bufV,128,(w+4)*16,a1,gb);
  }
  __syncthreads();
  for (int i=tid;i<16*192;i+=128){ // ve = R @ v[src], gated [go,gp,go]
    int r=i/192, q=i%192, j=q>>6, c=q&63;
    const float* vs = v + (size_t)ssrc[r]*192;
    float val = Rm[r*9+j*3+0]*vs[c] + Rm[r*9+j*3+1]*vs[64+c] + Rm[r*9+j*3+2]*vs[128+c];
    float gate = (j==1) ? bufV[r*128 + c] : bufV[r*128 + 64 + c];
    vev[i] = val*gate;
  }
  __syncthreads();
  for (int i=tid;i<16*192;i+=128){ // vm = R^T @ ve, scatter * mean(attn)
    int r=i/192, q=i%192, ii=q>>6, c=q&63;
    float vm = Rm[r*9+0+ii]*vev[r*192+c] + Rm[r*9+3+ii]*vev[r*192+64+c] + Rm[r*9+6+ii]*vev[r*192+128+c];
    atomicAdd(&ovacc[(size_t)sdst[r]*192 + ii*64 + c], vm*awv[r]);
  }
}

// ---- pass D: node-side oW/vmW projections + gated FFN -----------------------
__global__ __launch_bounds__(128) void k_nodeD(
    const float* __restrict__ osacc, const float* __restrict__ ovacc,
    const unsigned int* __restrict__ oWp, const float* __restrict__ ob,
    const unsigned int* __restrict__ vmWp,
    const unsigned int* __restrict__ fW1p, const float* __restrict__ fb1,
    const unsigned int* __restrict__ fW2p, const float* __restrict__ fb2,
    const unsigned int* __restrict__ fvWp,
    float* __restrict__ s, float* __restrict__ v)
{
  __shared__ unsigned int bA2[16*64];   // osacc bf16, later f1 bf16
  __shared__ unsigned int bO2[16*96];   // ovacc bf16
  __shared__ unsigned int bC2[16*64];   // cat(s,vn) bf16
  __shared__ float sloc[16*64];
  __shared__ float vloc[16*192];
  __shared__ float bF[16*128];          // f2 (f32)
  const int tid=threadIdx.x, w=tid>>5;
  const int n0=blockIdx.x*16;
  for (int i=tid;i<16*64;i+=128){
    int r=i>>6,c2=i&63;
    const float* p=osacc+(size_t)(n0+r)*128+2*c2;
    bA2[i]=pk2(p[0],p[1]);
  }
  for (int i=tid;i<16*96;i+=128){
    int r=i/96,c2=i%96;
    const float* p=ovacc+(size_t)(n0+r)*192+2*c2;
    bO2[i]=pk2(p[0],p[1]);
  }
  for (int i=tid;i<16*192;i+=128) vloc[i]=v[(size_t)n0*192+i];
  for (int i=tid;i<16*64;i+=128)  sloc[i]=s[(size_t)n0*64+i];
  __syncthreads();
  { v8f acc=gemmH(bA2,128,oWp,128,w*16); epi<0,true>(sloc,64,w*16,acc,ob); }
  for (int dd=0;dd<3;++dd){
    v8f acc=gemmH(bO2+dd*32,192,vmWp,64,w*16);
    epi<0,true>(vloc+dd*64,192,w*16,acc,nullptr);
  }
  __syncthreads();
  for (int i=tid;i<16*32;i+=128){
    int r=i>>5,c2=i&31; int c=2*c2;
    bC2[r*64+c2]=pk2(sloc[r*64+c],sloc[r*64+c+1]);
    float a0=vloc[r*192+c],  b0=vloc[r*192+64+c],  d0=vloc[r*192+128+c];
    float a1=vloc[r*192+c+1],b1=vloc[r*192+64+c+1],d1=vloc[r*192+128+c+1];
    bC2[r*64+32+c2]=pk2(sqrtf(a0*a0+b0*b0+d0*d0), sqrtf(a1*a1+b1*b1+d1*d1));
  }
  __syncthreads();
  { v8f a0=VZERO, a1=VZERO;
    gemmH2(bC2,128,fW1p,128,w*16,(w+4)*16,a0,a1);
    epiH<1>((__bf16*)bA2,128,w*16,a0,fb1);
    epiH<1>((__bf16*)bA2,128,(w+4)*16,a1,fb1);
  }
  __syncthreads();
  { v8f a0=VZERO, a1=VZERO;
    gemmH2(bA2,128,fW2p,128,w*16,(w+4)*16,a0,a1);
    epi<0,false>(bF,128,w*16,a0,fb2);
    epi<0,false>(bF,128,(w+4)*16,a1,fb2);
  }
  __syncthreads();
  for (int i=tid;i<16*64;i+=128){
    int r=i>>6,c=i&63;
    s[(size_t)n0*64+i] = sloc[i] + bF[r*128+c];
  }
  v8f va0=gemmF(vloc+0,  192,fvWp,64,w*16);
  v8f va1=gemmF(vloc+64, 192,fvWp,64,w*16);
  v8f va2=gemmF(vloc+128,192,fvWp,64,w*16);
  __syncthreads();
  {
    int lane=tid&31, cn=lane&15, mo=(lane>>4)*8;
#pragma unroll
    for (int r=0;r<8;++r){
      int m=mo+r, c=w*16+cn;
      float gate=bF[m*128+64+c];
      v[(size_t)(n0+m)*192 + c]       = vloc[m*192+c]     + va0[r]*gate;
      v[(size_t)(n0+m)*192 + 64 + c]  = vloc[m*192+64+c]  + va1[r]*gate;
      v[(size_t)(n0+m)*192 + 128 + c] = vloc[m*192+128+c] + va2[r]*gate;
    }
  }
}

// ---- node head helpers ------------------------------------------------------
__device__ __forceinline__ void xproj_st(float* catn, const float* nodef, const float* xb,
                                         const float* keepf, int n0, int col0, v8f acc){
  int lane=threadIdx.x&31, cn=lane&15, mo=(lane>>4)*8, col=col0+cn;
#pragma unroll
  for (int r=0;r<8;++r){
    int m=mo+r;
    catn[m*320+col] = nodef[(size_t)(n0+m)*256+col] + (acc[r]+xb[col])*keepf[m];
  }
}
__device__ __forceinline__ void h2_st(float* catn, const float* ntb2, int col0, v8f acc){
  int lane=threadIdx.x&31, cn=lane&15, mo=(lane>>4)*8, col=col0+cn;
#pragma unroll
  for (int r=0;r<8;++r){
    float x=acc[r]+ntb2[col];
    if (col0<256) catn[(mo+r)*320+col]+=x; else catn[(mo+r)*320+col]=x;
  }
}

// ---- node head: LN, node transition, backbone update, embed -----------------
__global__ __launch_bounds__(128) void k_post(
    const float* __restrict__ s, const float* __restrict__ v,
    const float* __restrict__ nodef,
    const float* __restrict__ qin, const float* __restrict__ Xin,
    const int* __restrict__ xmask, const int* __restrict__ nmask,
    const unsigned int* __restrict__ xWp, const float* __restrict__ xb,
    const float* __restrict__ ln1g, const float* __restrict__ ln1b,
    const unsigned int* __restrict__ ntW1p, const float* __restrict__ ntb1,
    const unsigned int* __restrict__ ntW2p, const float* __restrict__ ntb2,
    const float* __restrict__ ntlng, const float* __restrict__ ntlnb,
    const float* __restrict__ ntvW,
    const float* __restrict__ qW, const float* __restrict__ qb,
    const float* __restrict__ tW, const float* __restrict__ tb,
    const float* __restrict__ vw,
    float* __restrict__ o_nf, float* __restrict__ o_q,
    float* __restrict__ o_t, float* __restrict__ o_emb)
{
  __shared__ float sloc[16*64];
  __shared__ float vloc[16*192];
  __shared__ float catn[16*320];   // [nf(256) | vn(64)], later [nf0+h2 | gates]
  __shared__ unsigned int h1u[16*128]; // relu(h1) bf16 pairs
  __shared__ float nv2[16*192];
  __shared__ float keepf[16], nmf[16];
  __shared__ float mu1[16], rs1[16], mu2[16], rs2[16];
  const int tid=threadIdx.x, w=tid>>5;
  const int n0=blockIdx.x*16;
  for (int i=tid;i<16*64;i+=128){
    int r=i>>6, c=i&63;
    float x=s[(size_t)n0*64+i];
    sloc[i]=x;
    o_emb[(size_t)(n0+r)*256 + c] = x;
  }
  for (int i=tid;i<16*192;i+=128){
    int r=i/192, q=i%192;
    float x=v[(size_t)n0*192+i];
    vloc[i]=x;
    o_emb[(size_t)(n0+r)*256 + 64 + q] = x;
  }
  if (tid<16){ keepf[tid]=1.f-(float)xmask[n0+tid]; nmf[tid]=(float)nmask[n0+tid]; }
  __syncthreads();
  { v8f a0=VZERO,a1=VZERO,a2=VZERO,a3=VZERO;
    gemmF2(sloc,64,xWp,64,w*16,(w+4)*16,a0,a1);
    gemmF2(sloc,64,xWp,64,(w+8)*16,(w+12)*16,a2,a3);
    xproj_st(catn,nodef,xb,keepf,n0,w*16,a0);
    xproj_st(catn,nodef,xb,keepf,n0,(w+4)*16,a1);
    xproj_st(catn,nodef,xb,keepf,n0,(w+8)*16,a2);
    xproj_st(catn,nodef,xb,keepf,n0,(w+12)*16,a3);
  }
  __syncthreads();
  if (tid<16) rowstats(&catn[tid*320], 256, mu1[tid], rs1[tid]);
  __syncthreads();
  for (int i=tid;i<16*256;i+=128){
    int r=i>>8, c=i&255;
    catn[r*320+c]=(catn[r*320+c]-mu1[r])*rs1[r]*ln1g[c]+ln1b[c];
  }
  for (int i=tid;i<16*64;i+=128){
    int r=i>>6, c=i&63;
    float a=vloc[r*192+c], b=vloc[r*192+64+c], d2=vloc[r*192+128+c];
    catn[r*320+256+c]=sqrtf(a*a+b*b+d2*d2);
  }
  __syncthreads();
  { v8f a0=VZERO,a1=VZERO,a2=VZERO,a3=VZERO;
    gemmF2(catn,320,ntW1p,320,w*16,(w+4)*16,a0,a1);
    gemmF2(catn,320,ntW1p,320,(w+8)*16,(w+12)*16,a2,a3);
    epiH<2>((__bf16*)h1u,256,w*16,a0,ntb1);
    epiH<2>((__bf16*)h1u,256,(w+4)*16,a1,ntb1);
    epiH<2>((__bf16*)h1u,256,(w+8)*16,a2,ntb1);
    epiH<2>((__bf16*)h1u,256,(w+12)*16,a3,ntb1);
  }
  __syncthreads();
  { v8f a0=VZERO,a1=VZERO,a2=VZERO,a3=VZERO;
    gemmH2(h1u,256,ntW2p,256,w*16,(w+4)*16,a0,a1);
    gemmH2(h1u,256,ntW2p,256,(w+8)*16,(w+12)*16,a2,a3);
    v8f a4=gemmH(h1u,256,ntW2p,256,(w+16)*16);
    h2_st(catn,ntb2,w*16,a0);
    h2_st(catn,ntb2,(w+4)*16,a1);
    h2_st(catn,ntb2,(w+8)*16,a2);
    h2_st(catn,ntb2,(w+12)*16,a3);
    h2_st(catn,ntb2,(w+16)*16,a4);
  }
  __syncthreads();
  if (tid<16) rowstats(&catn[tid*320], 256, mu2[tid], rs2[tid]);
  __syncthreads();
  for (int i=tid;i<16*256;i+=128){
    int r=i>>8, c=i&255;
    float val=(catn[r*320+c]-mu2[r])*rs2[r]*ntlng[c]+ntlnb[c];
    val*=keepf[r];
    catn[r*320+c]=val;
    o_nf[(size_t)(n0+r)*256+c]=val;
  }
  __syncthreads();
  for (int i=tid;i<16*192;i+=128){
    int r=i/192, q=i%192, dd=q>>6, e2=q&63;
    float su=0.f;
    for (int c2=0;c2<64;++c2) su += vloc[r*192+dd*64+c2]*ntvW[c2*64+e2];
    nv2[r*192 + dd*64 + e2] = su * sigf(catn[r*320+256+e2]) * keepf[r];
  }
  __syncthreads();
  if (tid<16){
    int r=tid; int n=n0+r; float nm=nmf[r];
    float u[6];
    for (int j=0;j<3;++j){
      float sq=0.f, st=0.f;
      for (int c=0;c<256;++c){ float x=catn[r*320+c]; sq+=x*qW[c*3+j]; st+=x*tW[c*3+j]; }
      float ein=0.f;
      for (int e2=0;e2<64;++e2) ein += nv2[r*192 + j*64 + e2]*vw[e2];
      u[j]   = nm*(sq+qb[j]);
      u[3+j] = nm*(st+tb[j]+ein);
    }
    float w1=qin[n*4+0], x1=qin[n*4+1], y1=qin[n*4+2], z1=qin[n*4+3];
    float ow=w1        - x1*u[0] - y1*u[1] - z1*u[2];
    float ox=w1*u[0] + x1        + y1*u[2] - z1*u[1];
    float oy=w1*u[1] - x1*u[2] + y1        + z1*u[0];
    float oz=w1*u[2] + x1*u[1] - y1*u[0] + z1;
    float inv=rsqrtf(ow*ow+ox*ox+oy*oy+oz*oz);
    o_q[n*4+0]=ow*inv; o_q[n*4+1]=ox*inv; o_q[n*4+2]=oy*inv; o_q[n*4+3]=oz*inv;
    float R00=1.f-2.f*(y1*y1+z1*z1), R01=2.f*(x1*y1-w1*z1), R02=2.f*(x1*z1+w1*y1);
    float R10=2.f*(x1*y1+w1*z1), R11=1.f-2.f*(x1*x1+z1*z1), R12=2.f*(y1*z1-w1*x1);
    float R20=2.f*(x1*z1-w1*y1), R21=2.f*(y1*z1+w1*x1), R22=1.f-2.f*(x1*x1+y1*y1);
    o_t[n*3+0]=Xin[n*3+0]+R00*u[3]+R01*u[4]+R02*u[5];
    o_t[n*3+1]=Xin[n*3+1]+R10*u[3]+R11*u[4]+R12*u[5];
    o_t[n*3+2]=Xin[n*3+2]+R20*u[3]+R21*u[4]+R22*u[5];
  }
}

// ---- edge transition head ---------------------------------------------------
__global__ __launch_bounds__(128) void k_dproj(
    const float* __restrict__ nf, const unsigned int* __restrict__ dWp,
    const float* __restrict__ db, float* __restrict__ dout)
{
  __shared__ unsigned int buf2[16*128];
  const int tid=threadIdx.x, w=tid>>5;
  const int n0=blockIdx.x*16;
  for (int i=tid;i<16*128;i+=128){
    int r=i>>7,c2=i&127;
    const float* p=nf+(size_t)(n0+r)*256+2*c2;
    buf2[i]=pk2(p[0],p[1]);
  }
  __syncthreads();
  v8f acc=gemmH(buf2,256,dWp,256,w*16);
  int lane=tid&31, cn=lane&15, mo=(lane>>4)*8, col=w*16+cn;
#pragma unroll
  for (int r=0;r<8;++r) dout[(size_t)(n0+mo+r)*64+col]=acc[r]+db[col];
}

__global__ __launch_bounds__(128) void k_edgeT(
    const float* __restrict__ dfeat, const float* __restrict__ efeat,
    const int* __restrict__ src, const int* __restrict__ dst,
    const unsigned int* __restrict__ W1p, const float* __restrict__ b1,
    const unsigned int* __restrict__ W2p, const float* __restrict__ b2,
    const float* __restrict__ lng, const float* __restrict__ lnb,
    float* __restrict__ out)
{
  __shared__ float cat[16*256];        // f32: residual e needed exactly
  __shared__ unsigned int hh2[16*64];  // relu(h) bf16 pairs
  __shared__ float oo[16*128];
  __shared__ float mu[16], rs[16];
  const int tid=threadIdx.x, w=tid>>5;
  const int e0=blockIdx.x*16;
  if (blockIdx.x+1 < (int)gridDim.x)
    __builtin_prefetch(efeat + (size_t)(e0+16)*128 + tid*16, 0, 1);
  for (int i=tid;i<16*64;i+=128){
    int r=i>>6,c=i&63;
    cat[r*256+c]    = dfeat[(size_t)src[e0+r]*64+c];
    cat[r*256+64+c] = dfeat[(size_t)dst[e0+r]*64+c];
  }
  for (int i=tid;i<16*128;i+=128){
    int r=i>>7,c=i&127;
    cat[r*256+128+c]=efeat[(size_t)(e0+r)*128+c];
  }
  __syncthreads();
  { v8f a0=VZERO, a1=VZERO;
    gemmF2(cat,256,W1p,256,w*16,(w+4)*16,a0,a1);
    epiH<2>((__bf16*)hh2,128,w*16,a0,b1);
    epiH<2>((__bf16*)hh2,128,(w+4)*16,a1,b1);
  }
  __syncthreads();
  { v8f a0=VZERO, a1=VZERO;
    gemmH2(hh2,128,W2p,128,w*16,(w+4)*16,a0,a1);
    int lane=tid&31, cn=lane&15, mo=(lane>>4)*8;
#pragma unroll
    for (int r=0;r<8;++r){
      int m=mo+r;
      int c0=w*16+cn, c1=(w+4)*16+cn;
      oo[m*128+c0]=cat[m*256+128+c0]+a0[r]+b2[c0];
      oo[m*128+c1]=cat[m*256+128+c1]+a1[r]+b2[c1];
    }
  }
  __syncthreads();
  if (tid<16) rowstats(&oo[tid*128], 128, mu[tid], rs[tid]);
  __syncthreads();
  for (int i=tid;i<16*128;i+=128){
    int r=i>>7,c=i&127;
    out[(size_t)(e0+r)*128+c]=(oo[i]-mu[r])*rs[r]*lng[c]+lnb[c];
  }
}

// -----------------------------------------------------------------------------
extern "C" void kernel_launch(void* const* d_in, const int* in_sizes, int n_in,
                              void* d_out, int out_size, void* d_ws, size_t ws_size,
                              hipStream_t stream)
{
  (void)in_sizes; (void)n_in; (void)out_size; (void)ws_size;
#define PW(i) ((const float*)d_in[(i)])
  const float* nodef =(const float*)d_in[0];
  const float* qin   =(const float*)d_in[1];
  const float* Xin   =(const float*)d_in[2];
  const float* efeat =(const float*)d_in[3];
  const int*   eidx  =(const int*)d_in[4];
  const float* sefeat=(const float*)d_in[5];
  const int*   seidx =(const int*)d_in[6];
  const int*   nmask =(const int*)d_in[7];
  const int*   xmask =(const int*)d_in[8];
  const float* so3   =(const float*)d_in[9];
  // params flattened in sorted-key (jax tree) order
  const float *qW=PW(10),*qb=PW(11),*tW=PW(12),*tb=PW(13),*vw=PW(14);
  const float *cW=PW(15),*cb=PW(16);
  const float *etW1=PW(17),*etW2=PW(18),*etb1=PW(19),*etb2=PW(20),*etdW=PW(21),*etdb=PW(22),*etlnb=PW(23),*etlng=PW(24);
  const float *ln1b=PW(25),*ln1g=PW(26);
  const float *ntW1=PW(27),*ntW2=PW(28),*ntb1=PW(29),*ntb2=PW(30),*ntlnb=PW(31),*ntlng=PW(32),*ntvW=PW(33);
  const int SQ=34, SP=62; // per-block: +0 aW,+1 ab,+2 eW1,+3 eW2,+4 eb1,+5 eb2,
                          // +6 fW1,+7 fW2,+8 fb1,+9 fb2,+10 fvW,+11 gW,+12 gb,
                          // +13 mW,+14 mb,+15 oW,+16 ob,+17 vW,+18 vb,+19 vmW
  const float *stW1=PW(54),*stW2=PW(55),*stb1=PW(56),*stb2=PW(57),*stdW=PW(58),*stdb=PW(59),*stlnb=PW(60),*stlng=PW(61);
  const float *xW=PW(82),*xb=PW(83);

  char* wsb=(char*)d_ws;
  float* s_ws  =(float*)wsb; wsb += (size_t)NND*64*4;
  float* v_ws  =(float*)wsb; wsb += (size_t)NND*192*4;
  float* dbuf  =(float*)wsb; wsb += (size_t)NND*64*4;
  float* zacc  =(float*)wsb;            // contiguous zero region: osacc|ovacc|ssum|nmax
  float* osacc =(float*)wsb; wsb += (size_t)NND*128*4;
  float* ovacc =(float*)wsb; wsb += (size_t)NND*192*4;
  float* ssum  =(float*)wsb; wsb += (size_t)NND*8*4;
  unsigned int* nmax=(unsigned int*)wsb; wsb += (size_t)NND*8*4;
  float* logits=(float*)wsb; wsb += (size_t)NE*8*4;
  float* expw  =(float*)wsb; wsb += (size_t)NE*8*4;
  unsigned int* hbufu=(unsigned int*)wsb; wsb += (size_t)NE*64*4;
  unsigned int* pcur=(unsigned int*)wsb;  // packed bf16 weights (~1.25 MB)

  auto PACK=[&](const float* W, int K, int N)->const unsigned int*{
    unsigned int* dst=pcur; int total=(K/2)*N; pcur+=total;
    k_packw<<<(total+255)/256,256,0,stream>>>(W,dst,total,K,N);
    return dst;
  };
  const unsigned int* cWp  =PACK(cW,320,64);
  const unsigned int* xWp  =PACK(xW,64,256);
  const unsigned int* ntW1p=PACK(ntW1,320,256);
  const unsigned int* ntW2p=PACK(ntW2,256,320);
  const unsigned int* etW1p=PACK(etW1,256,128);
  const unsigned int* etW2p=PACK(etW2,128,128);
  const unsigned int* etdWp=PACK(etdW,256,64);
  const unsigned int* stW1p=PACK(stW1,256,128);
  const unsigned int* stW2p=PACK(stW2,128,128);
  const unsigned int* stdWp=PACK(stdW,256,64);
  struct BPk { const unsigned int *eW1,*eW2,*mW,*vW,*gW,*oW,*vmW,*fW1,*fW2,*fvW; };
  BPk bq, bp;
  {
    bq.eW1=PACK(PW(SQ+2),128,128); bq.eW2=PACK(PW(SQ+3),128,128);
    bq.mW =PACK(PW(SQ+13),256,128);
    bq.vW =PACK(PW(SQ+17),128,128); bq.gW =PACK(PW(SQ+11),128,128);
    bq.oW =PACK(PW(SQ+15),128,64);  bq.vmW=PACK(PW(SQ+19),64,64);
    bq.fW1=PACK(PW(SQ+6),128,128);  bq.fW2=PACK(PW(SQ+7),128,128);
    bq.fvW=PACK(PW(SQ+10),64,64);
    bp.eW1=PACK(PW(SP+2),128,128); bp.eW2=PACK(PW(SP+3),128,128);
    bp.mW =PACK(PW(SP+13),256,128);
    bp.vW =PACK(PW(SP+17),128,128); bp.gW =PACK(PW(SP+11),128,128);
    bp.oW =PACK(PW(SP+15),128,64);  bp.vmW=PACK(PW(SP+19),64,64);
    bp.fW1=PACK(PW(SP+6),128,128);  bp.fW2=PACK(PW(SP+7),128,128);
    bp.fvW=PACK(PW(SP+10),64,64);
  }

  float* out=(float*)d_out;
  float* o_nf = out;
  float* o_q  = o_nf + (size_t)NND*256;
  float* o_t  = o_q  + (size_t)NND*4;
  float* o_ef = o_t  + (size_t)NND*3;
  float* o_sef= o_ef + (size_t)NE*128;
  float* o_emb= o_sef+ (size_t)NSE*128;

  const int ZN = NND*(128+192+8+8);

  k_init<<<NND/16,128,0,stream>>>(so3, nodef, cWp, cb, s_ws, v_ws);

  // ---- seq attention block ----
  k_fill<<<(ZN+255)/256,256,0,stream>>>(zacc, ZN);
  k_edgeA<<<NSE/16,128,0,stream>>>(sefeat, seidx, seidx+NSE, s_ws,
      bq.eW1,PW(SQ+4),bq.eW2,PW(SQ+5),bq.mW,PW(SQ+14),PW(SQ+0),PW(SQ+1),
      hbufu, logits, nmax);
  k_edgeB<<<(NSE*8+255)/256,256,0,stream>>>(logits, seidx+NSE, nmax, ssum, expw, NSE);
  k_edgeC<<<NSE/16,128,0,stream>>>(hbufu, expw, ssum, seidx, seidx+NSE, Xin, v_ws,
      bq.vW,PW(SQ+18),bq.gW,PW(SQ+12), osacc, ovacc);
  k_nodeD<<<NND/16,128,0,stream>>>(osacc, ovacc,
      bq.oW,PW(SQ+16),bq.vmW,bq.fW1,PW(SQ+8),bq.fW2,PW(SQ+9),bq.fvW,
      s_ws, v_ws);

  // ---- spatial attention block ----
  k_fill<<<(ZN+255)/256,256,0,stream>>>(zacc, ZN);
  k_edgeA<<<NE/16,128,0,stream>>>(efeat, eidx, eidx+NE, s_ws,
      bp.eW1,PW(SP+4),bp.eW2,PW(SP+5),bp.mW,PW(SP+14),PW(SP+0),PW(SP+1),
      hbufu, logits, nmax);
  k_edgeB<<<(NE*8+255)/256,256,0,stream>>>(logits, eidx+NE, nmax, ssum, expw, NE);
  k_edgeC<<<NE/16,128,0,stream>>>(hbufu, expw, ssum, eidx, eidx+NE, Xin, v_ws,
      bp.vW,PW(SP+18),bp.gW,PW(SP+12), osacc, ovacc);
  k_nodeD<<<NND/16,128,0,stream>>>(osacc, ovacc,
      bp.oW,PW(SP+16),bp.vmW,bp.fW1,PW(SP+8),bp.fW2,PW(SP+9),bp.fvW,
      s_ws, v_ws);

  // ---- node head (nf, quats, trans, embed) ----
  k_post<<<NND/16,128,0,stream>>>(s_ws, v_ws, nodef, qin, Xin, xmask, nmask,
      xWp, xb, ln1g, ln1b, ntW1p, ntb1, ntW2p, ntb2, ntlng, ntlnb, ntvW,
      qW, qb, tW, tb, vw, o_nf, o_q, o_t, o_emb);

  // ---- edge transitions ----
  k_dproj<<<NND/16,128,0,stream>>>(o_nf, etdWp, etdb, dbuf);
  k_edgeT<<<NE/16,128,0,stream>>>(dbuf, efeat, eidx, eidx+NE,
      etW1p, etb1, etW2p, etb2, etlng, etlnb, o_ef);
  k_dproj<<<NND/16,128,0,stream>>>(o_nf, stdWp, stdb, dbuf);
  k_edgeT<<<NSE/16,128,0,stream>>>(dbuf, sefeat, seidx, seidx+NSE,
      stW1p, stb1, stW2p, stb2, stlng, stlnb, o_sef);
#undef PW
}
